// GAT_24592982737083
// MI455X (gfx1250) — matrix operand order repaired and checked by reading, not compiled, not run on hardware
//
#include <hip/hip_runtime.h>
#include <hip/hip_bf16.h>

// ---------------------------------------------------------------------------
// GAT forward for MI455X (gfx1250, wave32).
//   h = x @ W           -> v_wmma_f32_16x16x32_f16 (f16 in, f32 accum)
//   alpha_src/dst       -> wave-per-node dot + shfl reduction
//   segment softmax     -> 3 edge passes with global atomics (L2-resident)
//   scatter-add         -> wave-per-edge, coalesced float4, atomic f32 adds
// ---------------------------------------------------------------------------

#define NN        100000     // nodes
#define EE        3200000    // edges
#define FIN       128        // input features
#define FTOT      256        // HEADS * F_OUT
#define NEG_SLOPE 0.2f

typedef __attribute__((ext_vector_type(16))) _Float16 v16h;
typedef __attribute__((ext_vector_type(8)))  _Float16 v8h;
typedef __attribute__((ext_vector_type(8)))  float    v8f;

// Monotonic float<->uint encoding so atomicMax(u32) == float max.
__device__ __forceinline__ unsigned encf(float f) {
  unsigned b = __float_as_uint(f);
  return (b & 0x80000000u) ? ~b : (b | 0x80000000u);
}
__device__ __forceinline__ float decf(unsigned u) {
  unsigned b = (u & 0x80000000u) ? (u & 0x7FFFFFFFu) : ~u;
  return __uint_as_float(b);
}
__device__ __forceinline__ float leaky(float v) {
  return v > 0.0f ? v : NEG_SLOPE * v;
}

// ------------------------- conversion kernels ------------------------------

__global__ __launch_bounds__(256) void cvt_x_f16(const float* __restrict__ x,
                                                 _Float16* __restrict__ xh) {
  int i = blockIdx.x * 256 + threadIdx.x;           // NN*FIN elements
  if (i < NN * FIN) xh[i] = (_Float16)x[i];
}

// Wt[n][k] = (f16) W[k][n]  (W is [FIN, FTOT] row-major)
__global__ __launch_bounds__(256) void cvt_w_f16t(const float* __restrict__ w,
                                                  _Float16* __restrict__ wt) {
  int i = blockIdx.x * 256 + threadIdx.x;           // FTOT*FIN elements
  if (i < FTOT * FIN) {
    int n = i / FIN, k = i % FIN;
    wt[i] = (_Float16)w[k * FTOT + n];
  }
}

// ------------------------------ WMMA GEMM ----------------------------------
// Wave computes a 16x64 strip of h (one head's 64 features for 16 rows).
// grid = (ceil(6250/8), 4), block = 256 (8 waves).

__global__ __launch_bounds__(256) void gat_gemm_wmma(
    const _Float16* __restrict__ xh,   // [NN, FIN] row-major f16
    const _Float16* __restrict__ wt,   // [FTOT, FIN] (W transposed) f16
    float* __restrict__ h)             // [NN, FTOT] f32
{
  const int lane = threadIdx.x & 31;
  const int wave = threadIdx.x >> 5;
  const int rowGroup = blockIdx.x * 8 + wave;
  if (rowGroup >= NN / 16) return;                   // uniform per wave
  const int rowBase = rowGroup * 16;
  const int colBase = blockIdx.y * 64;

  const int m     = lane & 15;
  const int kaOff = (lane >> 4) * 8;                 // A frag K sub-offset
  const int kbOff = (lane >> 4) * 8;                 // B frag K sub-offset
  const _Float16* arow = xh + (size_t)(rowBase + m) * FIN;

  v8f acc[4] = {v8f{}, v8f{}, v8f{}, v8f{}};

#pragma unroll
  for (int kb = 0; kb < FIN; kb += 32) {
    v8h alo = *(const v8h*)(arow + kb + kaOff);
    v8h ahi = *(const v8h*)(arow + kb + 16 + kaOff);
    v16h a = __builtin_shufflevector(alo, ahi, 0, 1, 2, 3, 4, 5, 6, 7,
                                     8, 9, 10, 11, 12, 13, 14, 15);
#pragma unroll
    for (int t = 0; t < 4; ++t) {
      const _Float16* bcol =
          wt + (size_t)(colBase + t * 16 + m) * FIN + kb + kbOff;
      v8h blo = *(const v8h*)(bcol);
      v8h bhi = *(const v8h*)(bcol + 16);
      v16h b = __builtin_shufflevector(blo, bhi, 0, 1, 2, 3, 4, 5, 6, 7,
                                       8, 9, 10, 11, 12, 13, 14, 15);
      acc[t] = __builtin_amdgcn_wmma_f32_16x16x32_f16(
          /*neg_a=*/false, a, /*neg_b=*/false, b,
          /*c_mod=*/(short)0, acc[t], /*reuse_a=*/false, /*reuse_b=*/false);
    }
  }

  // C/D layout: VGPR r, lane L -> M = r + (L>>4)*8, N = L&15
  float* hb = h + (size_t)rowBase * FTOT + colBase;
  const int mrow = (lane >> 4) * 8;
#pragma unroll
  for (int t = 0; t < 4; ++t)
#pragma unroll
    for (int r = 0; r < 8; ++r)
      hb[(size_t)(mrow + r) * FTOT + t * 16 + (lane & 15)] = acc[t][r];
}

// -------------------------- per-node attention dots ------------------------
// Wave per node: lanes 0-15 reduce head (0,2), lanes 16-31 head (1,3).

__global__ __launch_bounds__(256) void gat_alpha(
    const float* __restrict__ h, const float* __restrict__ att_src,
    const float* __restrict__ att_dst, float* __restrict__ a_src,
    float* __restrict__ a_dst)
{
  const int lane = threadIdx.x & 31;
  const int wave = threadIdx.x >> 5;
  const int n = blockIdx.x * 8 + wave;
  if (n >= NN) return;

  const float* hrow = h + (size_t)n * FTOT;
  float4 c1 = *(const float4*)(hrow + 4 * lane);          // head  lane>>4
  float4 c2 = *(const float4*)(hrow + 128 + 4 * lane);    // head  2+(lane>>4)
  const int h1 = lane >> 4;
  const int cw = 4 * (lane & 15);

  const float* as1 = att_src + h1 * 64 + cw;
  const float* ad1 = att_dst + h1 * 64 + cw;
  const float* as2 = att_src + (h1 + 2) * 64 + cw;
  const float* ad2 = att_dst + (h1 + 2) * 64 + cw;

  float ps1 = c1.x * as1[0] + c1.y * as1[1] + c1.z * as1[2] + c1.w * as1[3];
  float pd1 = c1.x * ad1[0] + c1.y * ad1[1] + c1.z * ad1[2] + c1.w * ad1[3];
  float ps2 = c2.x * as2[0] + c2.y * as2[1] + c2.z * as2[2] + c2.w * as2[3];
  float pd2 = c2.x * ad2[0] + c2.y * ad2[1] + c2.z * ad2[2] + c2.w * ad2[3];

#pragma unroll
  for (int msk = 1; msk < 16; msk <<= 1) {
    ps1 += __shfl_xor(ps1, msk, 32);
    pd1 += __shfl_xor(pd1, msk, 32);
    ps2 += __shfl_xor(ps2, msk, 32);
    pd2 += __shfl_xor(pd2, msk, 32);
  }
  if ((lane & 15) == 0) {
    a_src[n * 4 + h1]     = ps1;
    a_src[n * 4 + h1 + 2] = ps2;
    a_dst[n * 4 + h1]     = pd1;
    a_dst[n * 4 + h1 + 2] = pd2;
  }
}

// ------------------------------- init --------------------------------------

__global__ __launch_bounds__(256) void gat_init(float* __restrict__ out,
                                                unsigned* __restrict__ nmax,
                                                float* __restrict__ nsum) {
  int i = blockIdx.x * 256 + threadIdx.x;            // NN*FTOT threads
  if (i < NN * FTOT) out[i] = 0.0f;
  if (i < NN * 4) {
    nmax[i] = 0x007FFFFFu;                           // encf(-inf)
    nsum[i] = 0.0f;
  }
}

// ----------------------- edge pass A: segment max --------------------------

__global__ __launch_bounds__(256) void gat_edge_max(
    const long long* __restrict__ ei, const float* __restrict__ a_src,
    const float* __restrict__ a_dst, unsigned* __restrict__ nmax)
{
  int e = blockIdx.x * 256 + threadIdx.x;
  if (e >= EE) return;
  long long s = ei[e], d = ei[(size_t)EE + e];
  float4 as = *(const float4*)(a_src + s * 4);
  float4 ad = *(const float4*)(a_dst + d * 4);
  float v[4] = {as.x + ad.x, as.y + ad.y, as.z + ad.z, as.w + ad.w};
#pragma unroll
  for (int hh = 0; hh < 4; ++hh)
    atomicMax(&nmax[d * 4 + hh], encf(leaky(v[hh])));
}

// ----------------------- edge pass B: segment sum --------------------------

__global__ __launch_bounds__(256) void gat_edge_sum(
    const long long* __restrict__ ei, const float* __restrict__ a_src,
    const float* __restrict__ a_dst, const unsigned* __restrict__ nmax,
    float* __restrict__ nsum)
{
  int e = blockIdx.x * 256 + threadIdx.x;
  if (e >= EE) return;
  long long s = ei[e], d = ei[(size_t)EE + e];
  float4 as = *(const float4*)(a_src + s * 4);
  float4 ad = *(const float4*)(a_dst + d * 4);
  float v[4] = {as.x + ad.x, as.y + ad.y, as.z + ad.z, as.w + ad.w};
#pragma unroll
  for (int hh = 0; hh < 4; ++hh) {
    float ex = expf(leaky(v[hh]) - decf(nmax[d * 4 + hh]));
    atomicAdd(&nsum[d * 4 + hh], ex);
  }
}

// ------------------ edge pass C: normalize + scatter-add -------------------
// Wave per edge: 256-float message, float4 per lane, coalesced row access.

__device__ __forceinline__ float edge_alpha(const float* a_src,
                                            const float* a_dst,
                                            const unsigned* nmax,
                                            const float* nsum, long long s,
                                            long long d, int hh) {
  float e0 = leaky(a_src[s * 4 + hh] + a_dst[d * 4 + hh]);
  float ex = expf(e0 - decf(nmax[d * 4 + hh]));
  return ex / (nsum[d * 4 + hh] + 1e-16f);
}

__global__ __launch_bounds__(256) void gat_scatter(
    const long long* __restrict__ ei, const float* __restrict__ h,
    const float* __restrict__ a_src, const float* __restrict__ a_dst,
    const unsigned* __restrict__ nmax, const float* __restrict__ nsum,
    float* __restrict__ out)
{
  const int lane = threadIdx.x & 31;
  const int wave = threadIdx.x >> 5;
  const long long e = (long long)blockIdx.x * 8 + wave;
  if (e >= EE) return;

  long long s = ei[e], d = ei[(size_t)EE + e];
  const float* hs = h + (size_t)s * FTOT;
  float4 v1 = *(const float4*)(hs + 4 * lane);         // head  lane>>4
  float4 v2 = *(const float4*)(hs + 128 + 4 * lane);   // head  2+(lane>>4)

  const int h1 = lane >> 4;
  float al1 = edge_alpha(a_src, a_dst, nmax, nsum, s, d, h1);
  float al2 = edge_alpha(a_src, a_dst, nmax, nsum, s, d, h1 + 2);

  float* od = out + (size_t)d * FTOT;
  atomicAdd(od + 4 * lane + 0, v1.x * al1);
  atomicAdd(od + 4 * lane + 1, v1.y * al1);
  atomicAdd(od + 4 * lane + 2, v1.z * al1);
  atomicAdd(od + 4 * lane + 3, v1.w * al1);
  atomicAdd(od + 128 + 4 * lane + 0, v2.x * al2);
  atomicAdd(od + 128 + 4 * lane + 1, v2.y * al2);
  atomicAdd(od + 128 + 4 * lane + 2, v2.z * al2);
  atomicAdd(od + 128 + 4 * lane + 3, v2.w * al2);
}

// ------------------------------- bias --------------------------------------

__global__ __launch_bounds__(256) void gat_bias(float* __restrict__ out,
                                                const float* __restrict__ bias) {
  int i = blockIdx.x * 256 + threadIdx.x;
  if (i < NN * FTOT) out[i] += bias[i & (FTOT - 1)];
}

// ------------------------------ launcher -----------------------------------

extern "C" void kernel_launch(void* const* d_in, const int* in_sizes, int n_in,
                              void* d_out, int out_size, void* d_ws,
                              size_t ws_size, hipStream_t stream) {
  (void)in_sizes; (void)n_in; (void)out_size; (void)ws_size;

  const float*     x       = (const float*)d_in[0];
  const long long* ei      = (const long long*)d_in[1];   // int64 edge_index
  const float*     W       = (const float*)d_in[2];
  const float*     att_src = (const float*)d_in[3];
  const float*     att_dst = (const float*)d_in[4];
  const float*     bias    = (const float*)d_in[5];
  float*           out     = (float*)d_out;

  // Workspace carve-up (all offsets 16B aligned). Total ~134.5 MB.
  char* ws = (char*)d_ws;
  _Float16* xh    = (_Float16*)(ws);                                  // 25.6 MB
  _Float16* wt    = (_Float16*)(ws + 25600000);                       // 64 KB
  float*    h     = (float*)   (ws + 25600000 + 65536);               // 102.4 MB
  float*    a_src = (float*)   (ws + 25600000 + 65536 + 102400000);   // 1.6 MB
  float*    a_dst = (float*)   ((char*)a_src + 1600000);              // 1.6 MB
  unsigned* nmax  = (unsigned*)((char*)a_dst + 1600000);              // 1.6 MB
  float*    nsum  = (float*)   ((char*)nmax + 1600000);               // 1.6 MB

  // 1) f16 conversion of x, f16 transpose of W
  cvt_x_f16<<<(NN * FIN + 255) / 256, 256, 0, stream>>>(x, xh);
  cvt_w_f16t<<<(FTOT * FIN + 255) / 256, 256, 0, stream>>>(W, wt);

  // 2) h = x @ W via WMMA. 6250 row-groups of 16, 8 waves/block, 4 col-groups.
  dim3 ggrid((NN / 16 + 7) / 8, 4);
  gat_gemm_wmma<<<ggrid, 256, 0, stream>>>(xh, wt, h);

  // 3) per-node attention coefficients
  gat_alpha<<<(NN + 7) / 8, 256, 0, stream>>>(h, att_src, att_dst, a_src, a_dst);

  // 4) init accumulators (d_out is poisoned by harness)
  gat_init<<<(NN * FTOT + 255) / 256, 256, 0, stream>>>(out, nmax, nsum);

  // 5) segment softmax: max, sum, then normalize+scatter
  gat_edge_max<<<(EE + 255) / 256, 256, 0, stream>>>(ei, a_src, a_dst, nmax);
  gat_edge_sum<<<(EE + 255) / 256, 256, 0, stream>>>(ei, a_src, a_dst, nmax, nsum);
  gat_scatter<<<(EE + 7) / 8, 256, 0, stream>>>(ei, h, a_src, a_dst, nmax, nsum, out);

  // 6) bias
  gat_bias<<<(NN * FTOT + 255) / 256, 256, 0, stream>>>(out, bias);
}
